// QMon_9775345566341
// MI455X (gfx1250) — matrix operand order repaired and checked by reading, not compiled, not run on hardware
//
#include <hip/hip_runtime.h>
#include <stdint.h>
#include <stddef.h>

// ---------------------------------------------------------------------------
// QMon (monDEQ) forward for gfx1250 (MI455X).
// Block-circulant DEQ operators materialized as dense 5184^2 f16 matrices
// (resident in 192MB L2); all heavy math is f16 WMMA GEMM with f32 accum.
// GEMM uses TDM (tensor_load_to_lds) double-buffered LDS staging.
// ---------------------------------------------------------------------------

#define F_DEQ   5184            // 64 channels * 9 * 9
#define NCH     64
#define NSP     9
#define BATCH   256
#define M_STR   0.1f
#define ALPHA_PR 1.0f
#define N_ITER  50
#define PI_F    3.14159265358979323846f

#define TM 64                   // block tile M (8 waves: 4 x 2 grid of 16x16)
#define TN 32                   // block tile N
#define TK 32                   // K step (one wmma_f32_16x16x32_f16)

typedef _Float16 f16;
typedef __attribute__((ext_vector_type(16))) _Float16 v16h;
typedef __attribute__((ext_vector_type(8)))  float    v8f;
typedef __attribute__((ext_vector_type(4)))  unsigned int v4u;
typedef __attribute__((ext_vector_type(4)))  int v4i;
typedef __attribute__((ext_vector_type(8)))  int v8i;

#ifndef __has_builtin
#define __has_builtin(x) 0
#endif
#if __has_builtin(__builtin_amdgcn_tensor_load_to_lds)
#define HAVE_TDM 1
#else
#define HAVE_TDM 0
#endif

union Frag16 { v16h v; uint32_t u[8]; };

// ------------------------- TDM 2D tile -> LDS -------------------------------
#if HAVE_TDM
// D# per CDNA5 ISA ch.8: group0 {count,lds_addr,global_addr,type=2},
// group1 {data_size=2B, tensor_dim0/1, tile_dim0/1, tensor_dim0_stride}.
__device__ __forceinline__ void tdm_load_tile(uint32_t lds_off, const f16* gsrc,
                                              int rows, int stride_elems) {
  uint64_t gaddr = (uint64_t)(uintptr_t)gsrc;
  v4u g0;
  g0[0] = 1u;                                        // count=1, user descriptor
  g0[1] = lds_off;                                   // lds_addr (bytes)
  g0[2] = (uint32_t)gaddr;                           // global_addr[31:0]
  g0[3] = (uint32_t)(gaddr >> 32) | (2u << 30);      // global_addr[56:32], type=2

  const uint64_t td0 = (uint64_t)TK;                 // tensor_dim0 (elems)
  const uint64_t td1 = (uint64_t)rows;               // tensor_dim1
  const uint64_t st  = (uint64_t)(uint32_t)stride_elems; // dim0 stride (elems)
  uint64_t q0 = (1ull << 16)                         // data_size = 1 -> 2 bytes
              | ((td0 & 0xFFFFull) << 48);           // tensor_dim0[15:0] @ bit48
  uint64_t q1 = (td0 >> 16)                          // tensor_dim0[31:16]
              | (td1 << 16)                          // tensor_dim1 @ bit80
              | ((uint64_t)TK << 48);                // tile_dim0 @ bit112
  uint64_t q2 = (uint64_t)rows                       // tile_dim1 @ bit128
              | ((st & 0xFFFFFFFFull) << 32);        // dim0_stride[31:0] @ bit160
  uint64_t q3 = (st >> 32);                          // dim0_stride[47:32]
  v8i g1;
  g1[0] = (int)q0; g1[1] = (int)(q0 >> 32);
  g1[2] = (int)q1; g1[3] = (int)(q1 >> 32);
  g1[4] = (int)q2; g1[5] = (int)(q2 >> 32);
  g1[6] = (int)q3; g1[7] = (int)(q3 >> 32);
  v4i z4 = {0, 0, 0, 0};
#if __clang_major__ >= 23
  v8i z8 = {0, 0, 0, 0, 0, 0, 0, 0};
  __builtin_amdgcn_tensor_load_to_lds(g0, g1, z4, z4, z8, 0);
#else
  __builtin_amdgcn_tensor_load_to_lds(g0, g1, z4, z4, 0);
#endif
}
__device__ __forceinline__ void tensor_wait() {
#if __has_builtin(__builtin_amdgcn_s_wait_tensorcnt)
  __builtin_amdgcn_s_wait_tensorcnt(0);
#else
  asm volatile("s_wait_tensorcnt 0x0" ::: "memory");
#endif
}
#endif // HAVE_TDM

// ------------------------- WMMA fragments from LDS --------------------------
// A: 16x32 f16 row-major panel. ISA 7.12.2: lanes 0-15 hold row M=lane,
// K pairs {0,2,4,6,16,18,20,22}; lanes 16-31 +8.
__device__ __forceinline__ v16h lds_a_frag(const f16* sm, int mOff) {
  const int lane = threadIdx.x & 31;
  const int m    = mOff + (lane & 15);
  const int half = lane >> 4;
  const uint32_t* p = reinterpret_cast<const uint32_t*>(sm + m * TK);
  Frag16 f;
#pragma unroll
  for (int v = 0; v < 8; ++v) {
    int k = ((v < 4) ? (2 * v) : (16 + 2 * (v - 4))) + 8 * half;
    f.u[v] = p[k >> 1];
  }
  return f.v;
}
// B stored transposed (N x K): lane holds col N=lane&15, K pairs {0,2,4,6,16,18,20,22} +8*half.
__device__ __forceinline__ v16h lds_b_frag(const f16* sm, int nOff) {
  const int lane = threadIdx.x & 31;
  const int n    = nOff + (lane & 15);
  const int half = lane >> 4;
  const uint32_t* p = reinterpret_cast<const uint32_t*>(sm + n * TK);
  Frag16 f;
#pragma unroll
  for (int v = 0; v < 8; ++v) {
    int k = ((v < 4) ? (2 * v) : (16 + 2 * (v - 4))) + 8 * half;
    f.u[v] = p[k >> 1];
  }
  return f.v;
}

// ------------------------------ generic GEMM --------------------------------
// 8 waves / block, 64x32 C macro-tile, one 16x16 tile per wave.
// K-panels staged into double-buffered LDS by the Tensor Data Mover (wave 0
// issues tensor_load_to_lds, waits TENSORcnt, workgroup barrier publishes).
// Epilogue modes:
//   0: f32 out = acc
//   1: f16 out = relu(acc + bias[n])
//   2: f16 out = (1-m)*delta(m==n) - g*acc        (forms W operator; g=sp[0])
//   3: f32 out = acc + bias[n]
__global__ void __launch_bounds__(256)
k_gemm(const f16* A, int lda, const f16* Bt, int ldb,
       void* outp, int N, int K, int mode,
       const float* __restrict__ bias, const float* __restrict__ sp)
{
  __shared__ f16 smA[2][TM * TK];     // 2 x 4KB
  __shared__ f16 smB[2][TN * TK];     // 2 x 2KB
  const int wave = threadIdx.x >> 5;
  const int lane = threadIdx.x & 31;
  const int blockTm = blockIdx.x * TM;
  const int blockTn = blockIdx.y * TN;
  const int mOff = 16 * (wave & 3);
  const int nOff = 16 * (wave >> 2);

#if HAVE_TDM
#define STAGE_ISSUE(buf, kk)                                                     \
  do {                                                                           \
    if (wave == 0) {                                                             \
      tdm_load_tile((uint32_t)(uintptr_t)&smA[buf][0],                           \
                    A + (size_t)blockTm * (size_t)lda + (kk), TM, lda);          \
      tdm_load_tile((uint32_t)(uintptr_t)&smB[buf][0],                           \
                    Bt + (size_t)blockTn * (size_t)ldb + (kk), TN, ldb);         \
    }                                                                            \
  } while (0)
#define STAGE_WAIT()                                                             \
  do { if (wave == 0) tensor_wait(); } while (0)
#else
#define STAGE_ISSUE(buf, kk)                                                     \
  do {                                                                           \
    uint32_t* dA = (uint32_t*)&smA[buf][0];                                      \
    uint32_t* dB = (uint32_t*)&smB[buf][0];                                      \
    for (int i = threadIdx.x; i < TM * TK / 2; i += 256) {                       \
      int r = i / (TK / 2), c = i % (TK / 2);                                    \
      dA[i] = ((const uint32_t*)(A + (size_t)(blockTm + r) * (size_t)lda + (kk)))[c]; \
    }                                                                            \
    for (int i = threadIdx.x; i < TN * TK / 2; i += 256) {                       \
      int r = i / (TK / 2), c = i % (TK / 2);                                    \
      dB[i] = ((const uint32_t*)(Bt + (size_t)(blockTn + r) * (size_t)ldb + (kk)))[c]; \
    }                                                                            \
  } while (0)
#define STAGE_WAIT() do { } while (0)
#endif

  v8f acc = {};
  STAGE_ISSUE(0, 0);
  STAGE_WAIT();
  __syncthreads();
  int cur = 0;
  for (int k = 0; k < K; k += TK) {
    if (k + TK < K) STAGE_ISSUE(cur ^ 1, k + TK);     // overlap TDM with WMMA
    v16h a = lds_a_frag(&smA[cur][0], mOff);
    v16h b = lds_b_frag(&smB[cur][0], nOff);
    acc = __builtin_amdgcn_wmma_f32_16x16x32_f16(false, a, false, b,
                                                 (short)0, acc, false, false);
    STAGE_WAIT();
    __syncthreads();
    cur ^= 1;
  }
#undef STAGE_ISSUE
#undef STAGE_WAIT

  const int tn = blockTn + nOff + (lane & 15);
  const int half = lane >> 4;
#pragma unroll
  for (int v = 0; v < 8; ++v) {
    const int m = blockTm + mOff + v + 8 * half;
    const size_t idx = (size_t)m * (size_t)N + tn;
    float r = acc[v];
    if (mode == 0) {
      ((float*)outp)[idx] = r;
    } else if (mode == 1) {
      r += bias[tn];
      ((f16*)outp)[idx] = (f16)fmaxf(r, 0.0f);
    } else if (mode == 2) {
      float d = (m == tn) ? (1.0f - M_STR) : 0.0f;
      ((f16*)outp)[idx] = (f16)(d - sp[0] * r);
    } else {
      ((float*)outp)[idx] = r + bias[tn];
    }
  }
}

// ------------------------------ utility kernels -----------------------------
__global__ void k_zero_u32(uint32_t* p, size_t n) {
  size_t i = (size_t)blockIdx.x * blockDim.x + threadIdx.x;
  size_t st = (size_t)gridDim.x * blockDim.x;
  for (; i < n; i += st) p[i] = 0u;
}
__global__ void k_cast_f16(const float* s, f16* d, size_t n) {
  size_t i = (size_t)blockIdx.x * blockDim.x + threadIdx.x;
  size_t st = (size_t)gridDim.x * blockDim.x;
  for (; i < n; i += st) d[i] = (f16)s[i];
}

// Frobenius norm of A_w (36864 floats), single block.
__global__ void k_norm(const float* A, float* nrm) {
  __shared__ float red[256];
  float s = 0.f;
  for (int i = threadIdx.x; i < NCH * NCH * 9; i += blockDim.x) { float v = A[i]; s += v * v; }
  red[threadIdx.x] = s; __syncthreads();
  for (int o = 128; o > 0; o >>= 1) { if (threadIdx.x < o) red[threadIdx.x] += red[threadIdx.x + o]; __syncthreads(); }
  if (threadIdx.x == 0) nrm[0] = sqrtf(red[0]);
}
__global__ void k_scaleA(const float* A, const float* nrm, float* An, int n) {
  int i = blockIdx.x * blockDim.x + threadIdx.x;
  if (i < n) An[i] = A[i] / nrm[0];
}

// Per-frequency Mmat = c*I + alpha*g * Af^H Af, Af = conj(DFT2(kimg)).
__global__ void k_freq(const float* __restrict__ An, const float* __restrict__ gp,
                       float* __restrict__ Mm)
{
  __shared__ float AfRe[NCH * NCH];
  __shared__ float AfIm[NCH * NCH];
  const int f  = blockIdx.x;
  const int fu = f / NSP, fv = f % NSP;
  for (int idx = threadIdx.x; idx < NCH * NCH; idx += blockDim.x) {
    const int o = idx >> 6, i = idx & 63;
    float re = 0.f, im = 0.f;
    for (int a = 0; a < 3; ++a)
      for (int b = 0; b < 3; ++b) {
        float w   = An[((o * NCH + i) * 3 + a) * 3 + b];
        float ang = -2.0f * PI_F * (float)(fu * (a - 1) + fv * (b - 1)) / (float)NSP;
        re += w * __cosf(ang); im += w * __sinf(ang);
      }
    AfRe[idx] = re; AfIm[idx] = -im;            // conj
  }
  __syncthreads();
  const float ag = ALPHA_PR * gp[0];
  const float c  = (1.0f + ALPHA_PR) - ALPHA_PR * (1.0f - M_STR);
  for (int idx = threadIdx.x; idx < NCH * NCH; idx += blockDim.x) {
    const int i = idx >> 6, j = idx & 63;
    float re = (i == j) ? c : 0.f, im = 0.f;
    for (int o = 0; o < NCH; ++o) {
      float ar = AfRe[o * NCH + i], ai = AfIm[o * NCH + i];
      float br = AfRe[o * NCH + j], bi = AfIm[o * NCH + j];
      re += ag * (ar * br + ai * bi);           // conj(a)*b
      im += ag * (ar * bi - ai * br);
    }
    float* out = Mm + ((size_t)f * (NCH * NCH) + idx) * 2;
    out[0] = re; out[1] = im;
  }
}

// In-place 64x64 complex Gauss-Jordan inversion per frequency (LDS resident).
__global__ void k_inv64(float* Mm) {
  __shared__ float Ar[NCH * NCH], Ai[NCH * NCH], Br[NCH * NCH], Bi[NCH * NCH];
  __shared__ float pr, pi;
  float* base = Mm + (size_t)blockIdx.x * (NCH * NCH) * 2;
  for (int idx = threadIdx.x; idx < NCH * NCH; idx += blockDim.x) {
    Ar[idx] = base[idx * 2]; Ai[idx] = base[idx * 2 + 1];
    Br[idx] = ((idx >> 6) == (idx & 63)) ? 1.f : 0.f; Bi[idx] = 0.f;
  }
  __syncthreads();
  for (int p = 0; p < NCH; ++p) {
    if (threadIdx.x == 0) {
      float dr = Ar[p * NCH + p], di = Ai[p * NCH + p];
      float inv = 1.0f / (dr * dr + di * di);
      pr = dr * inv; pi = -di * inv;
    }
    __syncthreads();
    for (int cc = threadIdx.x; cc < NCH; cc += blockDim.x) {
      float xr = Ar[p * NCH + cc], xi = Ai[p * NCH + cc];
      Ar[p * NCH + cc] = xr * pr - xi * pi; Ai[p * NCH + cc] = xr * pi + xi * pr;
      xr = Br[p * NCH + cc]; xi = Bi[p * NCH + cc];
      Br[p * NCH + cc] = xr * pr - xi * pi; Bi[p * NCH + cc] = xr * pi + xi * pr;
    }
    __syncthreads();
    for (int r = 0; r < NCH; ++r) {
      if (r == p) continue;
      float fr = Ar[r * NCH + p], fi2 = Ai[r * NCH + p];
      __syncthreads();
      for (int cc = threadIdx.x; cc < NCH; cc += blockDim.x) {
        float xr = Ar[p * NCH + cc], xi = Ai[p * NCH + cc];
        Ar[r * NCH + cc] -= fr * xr - fi2 * xi;
        Ai[r * NCH + cc] -= fr * xi + fi2 * xr;
        xr = Br[p * NCH + cc]; xi = Bi[p * NCH + cc];
        Br[r * NCH + cc] -= fr * xr - fi2 * xi;
        Bi[r * NCH + cc] -= fr * xi + fi2 * xr;
      }
      __syncthreads();
    }
  }
  for (int idx = threadIdx.x; idx < NCH * NCH; idx += blockDim.x) {
    base[idx * 2] = Br[idx]; base[idx * 2 + 1] = Bi[idx];
  }
}

// Spatial kernel of the inverse: Kinv[o,c,a,b] = (1/81) Re sum_f Vinv e^{+iw.d}
__global__ void k_ifft(const float* __restrict__ Vinv, float* __restrict__ Kinv) {
  size_t i = (size_t)blockIdx.x * blockDim.x + threadIdx.x;
  size_t st = (size_t)gridDim.x * blockDim.x;
  const size_t total = (size_t)NCH * NCH * 81;
  for (; i < total; i += st) {
    int oc = (int)(i / 81), d = (int)(i % 81);
    int a = d / NSP, b = d % NSP;
    float acc = 0.f;
    for (int f = 0; f < 81; ++f) {
      int fu = f / NSP, fv = f % NSP;
      float ang = 2.0f * PI_F * (float)(fu * a + fv * b) / (float)NSP;
      const float* v = Vinv + ((size_t)f * (NCH * NCH) + oc) * 2;
      acc += v[0] * __cosf(ang) - v[1] * __sinf(ang);
    }
    Kinv[i] = acc / 81.0f;
  }
}

// Dense circulant fill of the inverse operator (Bt layout: row fo, col fi).
__global__ void k_fill_V(const float* __restrict__ Kinv, f16* __restrict__ V) {
  size_t i = (size_t)blockIdx.x * blockDim.x + threadIdx.x;
  size_t st = (size_t)gridDim.x * blockDim.x;
  const size_t total = (size_t)F_DEQ * F_DEQ;
  for (; i < total; i += st) {
    int fo = (int)(i / F_DEQ), fi = (int)(i % F_DEQ);
    int o = fo / 81, p = fo % 81, c = fi / 81, q = fi % 81;
    int a = (p / NSP - q / NSP + NSP) % NSP;
    int b = (p % NSP - q % NSP + NSP) % NSP;
    V[i] = (f16)Kinv[(size_t)(o * NCH + c) * 81 + a * NSP + b];
  }
}

// Scatter a 3x3 circular-conv kernel into a dense 5184^2 operator.
// bt=0 -> row-major Mop (R[fi*F+fo]); bt=1 -> transposed Bt (R[fo*F+fi]).
__global__ void k_scatter_circ(const float* __restrict__ w, f16* __restrict__ dst, int bt) {
  size_t i = (size_t)blockIdx.x * blockDim.x + threadIdx.x;
  size_t st = (size_t)gridDim.x * blockDim.x;
  const size_t total = (size_t)NCH * NCH * 9 * 81;
  for (; i < total; i += st) {
    int p = (int)(i % 81); size_t r = i / 81;
    int dx = (int)(r % 3); r /= 3;
    int dy = (int)(r % 3); r /= 3;
    int ci = (int)(r % NCH); int o = (int)(r / NCH);
    float val = w[((o * NCH + ci) * 3 + dy) * 3 + dx];
    int py = p / NSP, px = p % NSP;
    int qy = (py + dy - 1 + NSP) % NSP, qx = (px + dx - 1 + NSP) % NSP;
    size_t fo = (size_t)o * 81 + p;
    size_t fi = (size_t)ci * 81 + qy * NSP + qx;
    if (bt) dst[fo * F_DEQ + fi] = (f16)val;
    else    dst[fi * F_DEQ + fo] = (f16)val;
  }
}

// ------------------------------ conv pre-net --------------------------------
__global__ void k_im2col1(const float* __restrict__ x, f16* __restrict__ out) {
  size_t i = (size_t)blockIdx.x * blockDim.x + threadIdx.x;
  size_t st = (size_t)gridDim.x * blockDim.x;
  const size_t total = (size_t)BATCH * 400 * 256;
  for (; i < total; i += st) {
    int kk = (int)(i % 256); size_t m = i / 256;
    int ox = (int)(m % 20); size_t r = m / 20;
    int oy = (int)(r % 20); int b = (int)(r / 20);
    int kx = kk & 7, ky = (kk >> 3) & 7, ci = kk >> 6;
    size_t src = (((size_t)b * 4 + ci) * 84 + (4 * oy + ky)) * 84 + (4 * ox + kx);
    out[i] = (f16)(x[src] * (1.0f / 255.0f));
  }
}
__global__ void k_im2col2(const f16* __restrict__ h1, f16* __restrict__ out) {
  size_t i = (size_t)blockIdx.x * blockDim.x + threadIdx.x;
  size_t st = (size_t)gridDim.x * blockDim.x;
  const size_t total = (size_t)BATCH * 81 * 512;
  for (; i < total; i += st) {
    int kk = (int)(i % 512); size_t m = i / 512;
    int ox = (int)(m % NSP); size_t r = m / NSP;
    int oy = (int)(r % NSP); int b = (int)(r / NSP);
    int kx = kk & 3, ky = (kk >> 2) & 3, ci = kk >> 4;
    size_t src = ((size_t)b * 400 + (2 * oy + ky) * 20 + (2 * ox + kx)) * 32 + ci;
    out[i] = h1[src];
  }
}
__global__ void k_pack_xpre(const f16* __restrict__ c2o, f16* __restrict__ X) {
  size_t i = (size_t)blockIdx.x * blockDim.x + threadIdx.x;
  size_t st = (size_t)gridDim.x * blockDim.x;
  const size_t total = (size_t)BATCH * F_DEQ;
  for (; i < total; i += st) {
    int f = (int)(i % F_DEQ); int b = (int)(i / F_DEQ);
    int c = f / 81, p = f % 81;
    X[i] = c2o[((size_t)b * 81 + p) * NCH + c];
  }
}
__global__ void k_bvecU(const float* __restrict__ Ub, float* __restrict__ bv) {
  int i = blockIdx.x * blockDim.x + threadIdx.x;
  if (i < F_DEQ) bv[i] = Ub[i / 81];
}

// ------------------------------ DEQ loop glue -------------------------------
__global__ void k_rhs(const float* __restrict__ z, const float* __restrict__ u,
                      const float* __restrict__ bias, float* __restrict__ t,
                      f16* __restrict__ rhs) {
  size_t i = (size_t)blockIdx.x * blockDim.x + threadIdx.x;
  size_t st = (size_t)gridDim.x * blockDim.x;
  const size_t total = (size_t)BATCH * F_DEQ;
  for (; i < total; i += st) {
    float u12 = 2.0f * z[i] - u[i];
    t[i]   = u12;
    rhs[i] = (f16)(u12 + ALPHA_PR * bias[i]);
  }
}
__global__ void k_update(const float* __restrict__ g, const float* __restrict__ t,
                         float* __restrict__ z, float* __restrict__ u,
                         f16* __restrict__ zh) {
  size_t i = (size_t)blockIdx.x * blockDim.x + threadIdx.x;
  size_t st = (size_t)gridDim.x * blockDim.x;
  const size_t total = (size_t)BATCH * F_DEQ;
  for (; i < total; i += st) {
    float un = 2.0f * g[i] - t[i];
    int p = (int)(i % 81);
    int py = p / NSP, px = p % NSP;
    bool interior = (py >= 1) && (py <= 7) && (px >= 1) && (px <= 7);
    float zn = interior ? fmaxf(un, 0.0f) : 0.0f;
    z[i] = zn; u[i] = un; zh[i] = (f16)zn;
  }
}
__global__ void k_finalize(const float* __restrict__ y, const float* __restrict__ bias,
                           f16* __restrict__ X1) {
  size_t i = (size_t)blockIdx.x * blockDim.x + threadIdx.x;
  size_t st = (size_t)gridDim.x * blockDim.x;
  const size_t total = (size_t)BATCH * F_DEQ;
  for (; i < total; i += st) {
    int f = (int)(i % F_DEQ); int b = (int)(i / F_DEQ);
    int c = f / 81, p = f % 81;
    int py = p / NSP, px = p % NSP;
    if (py >= 1 && py <= 7 && px >= 1 && px <= 7) {
      float v = fmaxf(y[i] + bias[i], 0.0f);
      X1[(size_t)b * 3136 + c * 49 + (py - 1) * 7 + (px - 1)] = (f16)v;
    }
  }
}
__global__ void k_pad_fc2(const float* __restrict__ w, f16* __restrict__ d) {
  int i = blockIdx.x * blockDim.x + threadIdx.x;
  if (i < 32 * 512) {
    int row = i / 512, col = i % 512;
    d[i] = (row < 18) ? (f16)w[row * 512 + col] : (f16)0.0f;
  }
}
__global__ void k_writeout(const float* __restrict__ fco, const float* __restrict__ b2,
                           float* __restrict__ out) {
  int i = blockIdx.x * blockDim.x + threadIdx.x;
  if (i < BATCH * 18) {
    int b = i / 18, j = i % 18;
    out[i] = fco[b * 32 + j] + b2[j];
  }
}

// ------------------------------ workspace layout ----------------------------
static constexpr size_t AL(size_t x) { return (x + 255) & ~(size_t)255; }
static constexpr size_t MATB  = (size_t)F_DEQ * F_DEQ * sizeof(f16);
static constexpr size_t o_RA  = 0;                                   // Ma / later U operator
static constexpr size_t o_W16 = AL(o_RA  + MATB);
static constexpr size_t o_V16 = AL(o_W16 + MATB);
static constexpr size_t o_IC  = AL(o_V16 + MATB);                    // im2col scratch
static constexpr size_t o_Mm  = AL(o_IC  + (size_t)102400 * 256 * 2);
static constexpr size_t o_An  = AL(o_Mm  + (size_t)81 * 4096 * 2 * 4);
static constexpr size_t o_nrm = AL(o_An  + (size_t)36864 * 4);
static constexpr size_t o_Ki  = AL(o_nrm + 256);
static constexpr size_t o_h1  = AL(o_Ki  + (size_t)331776 * 4);
static constexpr size_t o_c2o = AL(o_h1  + (size_t)102400 * 32 * 2);
static constexpr size_t o_Xp  = AL(o_c2o + (size_t)20736 * 64 * 2);
static constexpr size_t o_bias= AL(o_Xp  + (size_t)BATCH * F_DEQ * 2);
static constexpr size_t o_z   = AL(o_bias+ (size_t)BATCH * F_DEQ * 4);
static constexpr size_t o_u   = AL(o_z   + (size_t)BATCH * F_DEQ * 4);
static constexpr size_t o_t   = AL(o_u   + (size_t)BATCH * F_DEQ * 4);
static constexpr size_t o_g   = AL(o_t   + (size_t)BATCH * F_DEQ * 4);
static constexpr size_t o_y   = AL(o_g   + (size_t)BATCH * F_DEQ * 4);
static constexpr size_t o_rhs = AL(o_y   + (size_t)BATCH * F_DEQ * 4);
static constexpr size_t o_zh  = AL(o_rhs + (size_t)BATCH * F_DEQ * 2);
static constexpr size_t o_X1  = AL(o_zh  + (size_t)BATCH * F_DEQ * 2);
static constexpr size_t o_w1  = AL(o_X1  + (size_t)BATCH * 3136 * 2);
static constexpr size_t o_X2  = AL(o_w1  + (size_t)512 * 3136 * 2);
static constexpr size_t o_w2  = AL(o_X2  + (size_t)BATCH * 512 * 2);
static constexpr size_t o_c1w = AL(o_w2  + (size_t)32 * 512 * 2);
static constexpr size_t o_c2w = AL(o_c1w + (size_t)32 * 256 * 2);
static constexpr size_t o_bv  = AL(o_c2w + (size_t)64 * 512 * 2);
static constexpr size_t o_fco = AL(o_bv  + (size_t)F_DEQ * 4);

// ------------------------------ launcher ------------------------------------
extern "C" void kernel_launch(void* const* d_in, const int* in_sizes, int n_in,
                              void* d_out, int out_size, void* d_ws, size_t ws_size,
                              hipStream_t stream) {
  (void)in_sizes; (void)n_in; (void)out_size; (void)ws_size;
  const float* x       = (const float*)d_in[0];
  const float* c1w     = (const float*)d_in[1];
  const float* c1b     = (const float*)d_in[2];
  const float* c2wsrc  = (const float*)d_in[3];
  const float* c2b     = (const float*)d_in[4];
  const float* Uw      = (const float*)d_in[5];
  const float* Ub      = (const float*)d_in[6];
  const float* Aw      = (const float*)d_in[7];
  const float* gp      = (const float*)d_in[8];
  const float* fc1w    = (const float*)d_in[9];
  const float* fc1b    = (const float*)d_in[10];
  const float* fc2w    = (const float*)d_in[11];
  const float* fc2b    = (const float*)d_in[12];

  char* ws = (char*)d_ws;
  f16*   RA   = (f16*)(ws + o_RA);
  f16*   W16  = (f16*)(ws + o_W16);
  f16*   V16  = (f16*)(ws + o_V16);
  f16*   IC   = (f16*)(ws + o_IC);
  float* Mm   = (float*)(ws + o_Mm);
  float* An   = (float*)(ws + o_An);
  float* nrm  = (float*)(ws + o_nrm);
  float* Ki   = (float*)(ws + o_Ki);
  f16*   h1   = (f16*)(ws + o_h1);
  f16*   c2o  = (f16*)(ws + o_c2o);
  f16*   Xp   = (f16*)(ws + o_Xp);
  float* bias = (float*)(ws + o_bias);
  float* z32  = (float*)(ws + o_z);
  float* u32  = (float*)(ws + o_u);
  float* t32  = (float*)(ws + o_t);
  float* g32  = (float*)(ws + o_g);
  float* y32  = (float*)(ws + o_y);
  f16*   rhs  = (f16*)(ws + o_rhs);
  f16*   zh   = (f16*)(ws + o_zh);
  f16*   X1   = (f16*)(ws + o_X1);
  f16*   w1   = (f16*)(ws + o_w1);
  f16*   X2   = (f16*)(ws + o_X2);
  f16*   w2   = (f16*)(ws + o_w2);
  f16*   c1w16= (f16*)(ws + o_c1w);
  f16*   c2w16= (f16*)(ws + o_c2w);
  float* bv   = (float*)(ws + o_bv);
  float* fco  = (float*)(ws + o_fco);

  auto zero = [&](void* p, size_t bytes) {
    k_zero_u32<<<4096, 256, 0, stream>>>((uint32_t*)p, bytes / 4);
  };
  auto gemm = [&](const f16* A, int lda, const f16* Bt, int ldb, void* out,
                  int M, int N, int K, int mode, const float* bvec, const float* sp) {
    k_gemm<<<dim3(M / TM, N / TN), 256, 0, stream>>>(A, lda, Bt, ldb, out, N, K, mode, bvec, sp);
  };

  // ---- setup: frequency-domain inverse -> dense 5184^2 operator V ----
  k_norm<<<1, 256, 0, stream>>>(Aw, nrm);
  k_scaleA<<<(36864 + 255) / 256, 256, 0, stream>>>(Aw, nrm, An, 36864);
  k_freq<<<81, 256, 0, stream>>>(An, gp, Mm);
  k_inv64<<<81, 64, 0, stream>>>(Mm);
  k_ifft<<<2048, 256, 0, stream>>>(Mm, Ki);
  k_fill_V<<<8192, 256, 0, stream>>>(Ki, V16);

  // ---- setup: W = (1-m)I - g * Ma Ma^T  (RA serves as both A and Bt) ----
  zero(RA, MATB);
  k_scatter_circ<<<8192, 256, 0, stream>>>(An, RA, /*bt=*/0);
  gemm(RA, F_DEQ, RA, F_DEQ, W16, F_DEQ, F_DEQ, F_DEQ, /*mode=*/2, nullptr, gp);

  // ---- setup: U operator reuses RA region ----
  zero(RA, MATB);
  k_scatter_circ<<<8192, 256, 0, stream>>>(Uw, RA, /*bt=*/1);
  k_bvecU<<<(F_DEQ + 255) / 256, 256, 0, stream>>>(Ub, bv);

  // ---- weight casts ----
  k_cast_f16<<<512, 256, 0, stream>>>(c1w, c1w16, (size_t)32 * 256);
  k_cast_f16<<<512, 256, 0, stream>>>(c2wsrc, c2w16, (size_t)64 * 512);
  k_cast_f16<<<4096, 256, 0, stream>>>(fc1w, w1, (size_t)512 * 3136);
  k_pad_fc2<<<(32 * 512 + 255) / 256, 256, 0, stream>>>(fc2w, w2);

  // ---- conv pre-net (im2col + WMMA GEMM, relu+bias epilogue) ----
  k_im2col1<<<8192, 256, 0, stream>>>(x, IC);
  gemm(IC, 256, c1w16, 256, h1, 102400, 32, 256, /*mode=*/1, c1b, nullptr);   // (B*400)x32
  k_im2col2<<<8192, 256, 0, stream>>>(h1, IC);
  gemm(IC, 512, c2w16, 512, c2o, 20736, 64, 512, /*mode=*/1, c2b, nullptr);   // (B*81)x64
  k_pack_xpre<<<4096, 256, 0, stream>>>(c2o, Xp);

  // ---- bias = circ_conv(xpre, U) + U_b, as GEMM against U operator ----
  gemm(Xp, F_DEQ, RA, F_DEQ, bias, BATCH, F_DEQ, F_DEQ, /*mode=*/3, bv, nullptr);

  // ---- Peaceman-Rachford: 50 fixed iterations (MAX_ITER upper bound) ----
  zero(z32, (size_t)BATCH * F_DEQ * 4);
  zero(u32, (size_t)BATCH * F_DEQ * 4);
  for (int it = 0; it < N_ITER; ++it) {
    k_rhs<<<4096, 256, 0, stream>>>(z32, u32, bias, t32, rhs);
    gemm(rhs, F_DEQ, V16, F_DEQ, g32, BATCH, F_DEQ, F_DEQ, /*mode=*/0, nullptr, nullptr);
    k_update<<<4096, 256, 0, stream>>>(g32, t32, z32, u32, zh);
  }

  // ---- final pass: brelu(W z + bias), crop border, flatten ----
  gemm(zh, F_DEQ, W16, F_DEQ, y32, BATCH, F_DEQ, F_DEQ, /*mode=*/0, nullptr, nullptr);
  k_finalize<<<4096, 256, 0, stream>>>(y32, bias, X1);

  // ---- MLP head ----
  gemm(X1, 3136, w1, 3136, X2, BATCH, 512, 3136, /*mode=*/1, fc1b, nullptr);
  gemm(X2, 512, w2, 512, fco, BATCH, 32, 512, /*mode=*/0, nullptr, nullptr);
  k_writeout<<<(BATCH * 18 + 255) / 256, 256, 0, stream>>>(fco, fc2b, (float*)d_out);
}
